// CrossAttention_86002425135537
// MI455X (gfx1250) — matrix operand order repaired and checked by reading, not compiled, not run on hardware
//
#include <hip/hip_runtime.h>

// ---------------- problem constants ----------------
#define BATCH   4
#define SEQ     1024
#define HIDDEN  1024
#define NHEADS  16
#define HEADDIM 64
#define NPREV   4
#define EMB     1024   // NHEADS*HEADDIM

typedef __attribute__((ext_vector_type(16))) __bf16 v16bf;
typedef __attribute__((ext_vector_type(8)))  __bf16 v8bf;
typedef __attribute__((ext_vector_type(8)))  float  v8f;
typedef int v4i_vs __attribute__((vector_size(4 * sizeof(int))));

// ---------------- CDNA5 async global->LDS staging ----------------
#if defined(__has_builtin)
# if __has_builtin(__builtin_amdgcn_global_load_async_to_lds_b128)
#  define HAVE_ASYNC_LDS 1
# endif
#endif
#ifndef HAVE_ASYNC_LDS
# define HAVE_ASYNC_LDS 0
#endif

__device__ __forceinline__ void async_copy16(const void* g, void* l) {
#if HAVE_ASYNC_LDS
  __builtin_amdgcn_global_load_async_to_lds_b128(
      (__attribute__((address_space(1))) v4i_vs*)g,
      (__attribute__((address_space(3))) v4i_vs*)l,
      0, 0);
#else
  *(uint4*)l = *(const uint4*)g;   // fallback: stage through VGPRs
#endif
}

__device__ __forceinline__ void async_wait() {
#if HAVE_ASYNC_LDS
# if defined(__has_builtin) && __has_builtin(__builtin_amdgcn_s_wait_asynccnt)
  __builtin_amdgcn_s_wait_asynccnt(0);
# else
  asm volatile("s_wait_asynccnt 0" ::: "memory");
# endif
#endif
}

__device__ __forceinline__ v8f vzero8() { v8f v = {0.f,0.f,0.f,0.f,0.f,0.f,0.f,0.f}; return v; }

__device__ __forceinline__ v8f wmma_bf16(v16bf a, v16bf b, v8f c) {
  // D = A(16x32 bf16) * B(32x16 bf16) + C(16x16 f32)
  return __builtin_amdgcn_wmma_f32_16x16x32_bf16(false, a, false, b, (short)0, c, false, false);
}

// Build a v16bf fragment from two contiguous 8-half (16B) chunks.
__device__ __forceinline__ v16bf ld_frag2(const __bf16* p0, const __bf16* p1) {
  union { v16bf v; v8bf h[2]; } u;
  u.h[0] = *(const v8bf*)p0;
  u.h[1] = *(const v8bf*)p1;
  return u.v;
}

// ---------------- weight transpose + bf16 convert ----------------
// wt[n][k] = (bf16) w[k][n], both 1024x1024
__global__ __launch_bounds__(256)
void transpose_bf16_kernel(const float* __restrict__ w, __bf16* __restrict__ wt) {
  int idx = blockIdx.x * 256 + threadIdx.x;   // < 1024*1024
  int n = idx >> 10;
  int k = idx & 1023;
  wt[(size_t)n * 1024 + k] = (__bf16)w[(size_t)k * 1024 + n];
}

// ---------------- gating: pooled mean + softmax(wg) ----------------
__global__ __launch_bounds__(256)
void pool_kernel(const float* __restrict__ x, float* __restrict__ pooled) {
  int idx = blockIdx.x * 256 + threadIdx.x;   // < BATCH*HIDDEN = 4096
  int bb = idx >> 10;
  int hc = idx & 1023;
  float sum = 0.f;
  for (int s = 0; s < SEQ; ++s)
    sum += x[((size_t)bb * SEQ + s) * HIDDEN + hc];
  pooled[idx] = sum * (1.0f / (float)SEQ);
}

__global__ __launch_bounds__(512)
void gate_kernel(const float* __restrict__ pooled, const float* __restrict__ wg,
                 float* __restrict__ lwout) {
  __shared__ float logits[BATCH * NPREV];
  int t = threadIdx.x;            // 512 threads: 16 (b,p) pairs x 32 lanes
  int bp = t >> 5, lane = t & 31;
  int bb = bp >> 2, p = bp & 3;
  float sum = 0.f;
  for (int i = lane; i < HIDDEN; i += 32)
    sum += pooled[bb * HIDDEN + i] * wg[i * NPREV + p];
  for (int off = 16; off >= 1; off >>= 1) sum += __shfl_down(sum, off, 32);
  if (lane == 0) logits[bp] = sum;
  __syncthreads();
  if (t < BATCH) {
    float mx = -3.0e38f;
    for (int j = 0; j < NPREV; ++j) mx = fmaxf(mx, logits[t * NPREV + j]);
    float e[NPREV], den = 0.f;
    for (int j = 0; j < NPREV; ++j) { e[j] = __expf(logits[t * NPREV + j] - mx); den += e[j]; }
    for (int j = 0; j < NPREV; ++j) lwout[t * NPREV + j] = e[j] / den;
  }
}

// ---------------- WMMA GEMM: C[M,1024] = A[M,1024] @ WT^T + bias ----------------
// WT is [N=1024][K=1024] bf16 (pre-transposed weight).
// OUTMODE 0: f32 row-major [M,1024]
// OUTMODE 1: bf16 head layout      [bb, head, s, d]   (bb = row/SEQ)
// OUTMODE 2: bf16 head-T layout    [bb, head, d, s]   (for V, feeds P.V B-fragments)
template<int ABF16, int OUTMODE, int ROPE>
__global__ __launch_bounds__(256)
void proj_gemm(const void* __restrict__ Aptr, int M,
               const __bf16* __restrict__ WT,
               const float* __restrict__ bias,
               void* __restrict__ outp) {
  __shared__ __align__(16) __bf16 sA[128 * 40];  // 128 rows x 32 K (stride 40)
  __shared__ __align__(16) __bf16 sB[64 * 40];   // 64 cols  x 32 K (stride 40)

  const int tid = threadIdx.x;
  const int lane = tid & 31;
  const int w = tid >> 5;          // wave id 0..7 -> M strip
  const int lanelo = lane & 15;
  const int hi = lane >> 4;
  const int n0 = blockIdx.x * 64;
  const int m0 = blockIdx.y * 128;
  (void)M;

  v8f acc[4];
  #pragma unroll
  for (int j = 0; j < 4; ++j) acc[j] = vzero8();

  for (int k0 = 0; k0 < HIDDEN; k0 += 32) {
    // ---- stage B tile (64 cols x 32 K), async when available ----
    {
      int r = tid >> 2;
      int c = (tid & 3) * 8;
      async_copy16(&WT[(size_t)(n0 + r) * HIDDEN + k0 + c], &sB[r * 40 + c]);
    }
    // ---- stage A tile (128x32) into LDS as bf16 ----
    if constexpr (ABF16) {
      const __bf16* A = (const __bf16*)Aptr;
      #pragma unroll
      for (int i = 0; i < 2; ++i) {
        int e = tid + i * 256;          // 512 chunks of 8 halfs
        int r = e >> 2;
        int c = (e & 3) * 8;
        async_copy16(&A[(size_t)(m0 + r) * HIDDEN + k0 + c], &sA[r * 40 + c]);
      }
    } else {
      const float* A = (const float*)Aptr;
      #pragma unroll
      for (int i = 0; i < 4; ++i) {
        int e = tid + i * 256;          // 1024 chunks of 4 floats
        int r = e >> 3;
        int c = (e & 7) * 4;
        float4 f = *(const float4*)&A[(size_t)(m0 + r) * HIDDEN + k0 + c];
        __bf16* d = &sA[r * 40 + c];
        d[0] = (__bf16)f.x; d[1] = (__bf16)f.y; d[2] = (__bf16)f.z; d[3] = (__bf16)f.w;
      }
    }
    async_wait();
    __syncthreads();

    // A fragment: row = wave strip + lanelo; chunks at K = hi*8 and 16+hi*8
    const __bf16* arow = &sA[(w * 16 + lanelo) * 40 + hi * 8];
    v16bf af = ld_frag2(arow, arow + 16);
    #pragma unroll
    for (int j = 0; j < 4; ++j) {
      const __bf16* brow = &sB[(j * 16 + lanelo) * 40 + hi * 8];
      v16bf bf = ld_frag2(brow, brow + 16);
      acc[j] = wmma_bf16(af, bf, acc[j]);
    }
    __syncthreads();
  }

  // ---- epilogue: bias ----
  #pragma unroll
  for (int j = 0; j < 4; ++j) {
    float bj = bias[n0 + j * 16 + lanelo];
    #pragma unroll
    for (int r = 0; r < 8; ++r) acc[j][r] += bj;
  }

  // ---- fused RoPE (rotate-half over head dim 64; wave tile = one full head) ----
  if constexpr (ROPE) {
    #pragma unroll
    for (int j = 0; j < 2; ++j) {
      int d = j * 16 + lanelo;                       // d in [0,32)
      float invf = __expf(-(float)d * 0.2878231366242561f);  // ln(10000)/32
      #pragma unroll
      for (int r = 0; r < 8; ++r) {
        int mrow = m0 + w * 16 + r + hi * 8;
        int s = mrow & (SEQ - 1);
        float ang = (float)s * invf;
        float cs = __cosf(ang), sn = __sinf(ang);
        float x1 = acc[j][r], x2 = acc[j + 2][r];
        acc[j][r]     = x1 * cs - x2 * sn;
        acc[j + 2][r] = x1 * sn + x2 * cs;
      }
    }
  }

  // ---- store ----
  if constexpr (OUTMODE == 1) {
    __bf16* out = (__bf16*)outp;
    #pragma unroll
    for (int j = 0; j < 4; ++j) {
      int n = n0 + j * 16 + lanelo;
      int h = n >> 6, d = n & 63;
      #pragma unroll
      for (int r = 0; r < 8; ++r) {
        int mrow = m0 + w * 16 + r + hi * 8;
        int bb = mrow >> 10;                 // row / SEQ
        int s = mrow & (SEQ - 1);
        out[(((size_t)bb * NHEADS + h) * SEQ + s) * HEADDIM + d] = (__bf16)acc[j][r];
      }
    }
  } else if constexpr (OUTMODE == 2) {
    __bf16* out = (__bf16*)outp;
    #pragma unroll
    for (int j = 0; j < 4; ++j) {
      int n = n0 + j * 16 + lanelo;
      int h = n >> 6, d = n & 63;
      #pragma unroll
      for (int r = 0; r < 8; ++r) {
        int mrow = m0 + w * 16 + r + hi * 8;
        int bb = mrow >> 10;
        int s = mrow & (SEQ - 1);
        out[(((size_t)bb * NHEADS + h) * HEADDIM + d) * SEQ + s] = (__bf16)acc[j][r];
      }
    }
  } else {
    float* out = (float*)outp;
    #pragma unroll
    for (int j = 0; j < 4; ++j) {
      int n = n0 + j * 16 + lanelo;
      #pragma unroll
      for (int r = 0; r < 8; ++r) {
        int mrow = m0 + w * 16 + r + hi * 8;
        out[(size_t)mrow * 1024 + n] = acc[j][r];
      }
    }
  }
}

// ---------------- fused flash attention + layer gating ----------------
// grid = (S/128, NHEADS, BATCH); block = 256 (8 waves, each owns 16 q rows).
// Loops p = 0..NPREV-1 inside the block, accumulating gated output.
__global__ __launch_bounds__(256)
void attention_kernel(const __bf16* __restrict__ Q,     // [B,NH,S,HD]
                      const __bf16* __restrict__ K,     // [P,B,NH,S,HD]
                      const __bf16* __restrict__ Vt,    // [P,B,NH,HD,S]  (transposed)
                      const float*  __restrict__ layer_w, // [B,P]
                      __bf16* __restrict__ comb) {      // [B,S,EMB]
  __shared__ __align__(16) __bf16 sK[32 * 72];          // 32 keys x 64 d (stride 72)
  __shared__ __align__(16) __bf16 sV[64 * 40];          // 64 d x 32 keys (stride 40)
  __shared__ __align__(16) __bf16 sP[8 * 16 * 40];      // per-wave 16x32 P tile

  const int tid = threadIdx.x, lane = tid & 31, w = tid >> 5;
  const int lanelo = lane & 15, hi = lane >> 4;
  const int b = blockIdx.z, h = blockIdx.y;
  const int q0 = blockIdx.x * 128;
  const int qrow = q0 + w * 16 + lanelo;

  // Q fragments (two K=32 d-chunks), held in registers for the whole kernel.
  const size_t qbase = (((size_t)b * NHEADS + h) * SEQ + qrow) * HEADDIM;
  v16bf qf[2];
  qf[0] = ld_frag2(&Q[qbase + hi * 8],      &Q[qbase + 16 + hi * 8]);
  qf[1] = ld_frag2(&Q[qbase + 32 + hi * 8], &Q[qbase + 48 + hi * 8]);

  float lw[NPREV];
  #pragma unroll
  for (int p = 0; p < NPREV; ++p) lw[p] = layer_w[b * NPREV + p];

  v8f gacc[4];
  #pragma unroll
  for (int j = 0; j < 4; ++j) gacc[j] = vzero8();

  // per-thread staging coordinates
  const int kr_k = tid >> 3, c_k = (tid & 7) * 8;   // K tile:  32 rows x 64 halfs
  const int d_v  = tid >> 2, c_v = (tid & 3) * 8;   // Vt tile: 64 rows x 32 halfs

  for (int p = 0; p < NPREV; ++p) {
    const size_t kvbase = (((size_t)p * BATCH + b) * NHEADS + h) * (size_t)SEQ * HEADDIM;
    float rm[8], rl[8];
    #pragma unroll
    for (int r = 0; r < 8; ++r) { rm[r] = -3.0e38f; rl[r] = 0.f; }
    v8f oacc[4];
    #pragma unroll
    for (int j = 0; j < 4; ++j) oacc[j] = vzero8();

    for (int kt = 0; kt < SEQ / 32; ++kt) {
      const int k0 = kt * 32;
      __syncthreads();  // all waves done reading previous K/V tiles
      // stage K tile [key][d] and V^T tile [d][key], async when available
      async_copy16(&K [kvbase + (size_t)(k0 + kr_k) * HEADDIM + c_k], &sK[kr_k * 72 + c_k]);
      async_copy16(&Vt[kvbase + (size_t)d_v * SEQ + k0 + c_v],        &sV[d_v * 40 + c_v]);
      async_wait();
      __syncthreads();

      // scores: 16 q x 32 keys = 2 N-tiles x 2 K(d)-chunks
      v8f sacc[2];
      sacc[0] = vzero8(); sacc[1] = vzero8();
      #pragma unroll
      for (int dk = 0; dk < 2; ++dk) {
        #pragma unroll
        for (int j = 0; j < 2; ++j) {
          const __bf16* br = &sK[(j * 16 + lanelo) * 72 + dk * 32 + hi * 8];
          v16bf bf = ld_frag2(br, br + 16);
          sacc[j] = wmma_bf16(qf[dk], bf, sacc[j]);
        }
      }

      // online softmax (rows live across 16-lane groups in the C layout)
      __bf16* sPw = &sP[w * 16 * 40];
      #pragma unroll
      for (int r = 0; r < 8; ++r) {
        float s0 = sacc[0][r] * 0.125f;
        float s1 = sacc[1][r] * 0.125f;
        float mx = fmaxf(s0, s1);
        #pragma unroll
        for (int off = 8; off >= 1; off >>= 1) mx = fmaxf(mx, __shfl_xor(mx, off, 32));
        float nm = fmaxf(rm[r], mx);
        float corr = __expf(rm[r] - nm);
        float p0 = __expf(s0 - nm);
        float p1 = __expf(s1 - nm);
        float sum = p0 + p1;
        #pragma unroll
        for (int off = 8; off >= 1; off >>= 1) sum += __shfl_xor(sum, off, 32);
        rl[r] = rl[r] * corr + sum;
        rm[r] = nm;
        #pragma unroll
        for (int dt = 0; dt < 4; ++dt) oacc[dt][r] *= corr;
        int ml = r + hi * 8;
        sPw[ml * 40 + lanelo]      = (__bf16)p0;
        sPw[ml * 40 + 16 + lanelo] = (__bf16)p1;
      }
      // same-wave LDS RAW: make P stores visible before re-fragmenting
      asm volatile("s_wait_dscnt 0" ::: "memory");

      const __bf16* pr = &sPw[lanelo * 40 + hi * 8];
      v16bf pf = ld_frag2(pr, pr + 16);
      #pragma unroll
      for (int dt = 0; dt < 4; ++dt) {
        const __bf16* br = &sV[(dt * 16 + lanelo) * 40 + hi * 8];
        v16bf bf = ld_frag2(br, br + 16);
        oacc[dt] = wmma_bf16(pf, bf, oacc[dt]);
      }
    }

    // finalize this prev layer: normalize and apply gate weight
    float g = lw[p];
    #pragma unroll
    for (int dt = 0; dt < 4; ++dt) {
      #pragma unroll
      for (int r = 0; r < 8; ++r)
        gacc[dt][r] += g * (oacc[dt][r] / rl[r]);
    }
  }

  // store combined bf16 [B,S,EMB]
  #pragma unroll
  for (int dt = 0; dt < 4; ++dt) {
    int e = h * HEADDIM + dt * 16 + lanelo;
    #pragma unroll
    for (int r = 0; r < 8; ++r) {
      int s = q0 + w * 16 + r + hi * 8;
      comb[((size_t)b * SEQ + s) * EMB + e] = (__bf16)gacc[dt][r];
    }
  }
}

// ---------------- launch ----------------
extern "C" void kernel_launch(void* const* d_in, const int* in_sizes, int n_in,
                              void* d_out, int out_size, void* d_ws, size_t ws_size,
                              hipStream_t stream) {
  (void)in_sizes; (void)n_in; (void)out_size; (void)ws_size;
  const float* hidden = (const float*)d_in[0];
  const float* prev   = (const float*)d_in[1];
  const float* wq = (const float*)d_in[2];
  const float* bq = (const float*)d_in[3];
  const float* wk = (const float*)d_in[4];
  const float* bk = (const float*)d_in[5];
  const float* wv = (const float*)d_in[6];
  const float* bv = (const float*)d_in[7];
  const float* wo = (const float*)d_in[8];
  const float* bo = (const float*)d_in[9];
  const float* wg = (const float*)d_in[10];

  char* ws = (char*)d_ws;
  size_t off = 0;
  auto alloc = [&](size_t bytes) -> void* {
    void* p = ws + off;
    off = (off + bytes + 255) & ~(size_t)255;
    return p;
  };
  float*  layer_w = (float*)alloc(BATCH * NPREV * sizeof(float));
  float*  pooled  = (float*)alloc(BATCH * HIDDEN * sizeof(float));
  __bf16* wqT = (__bf16*)alloc((size_t)HIDDEN * EMB * 2);
  __bf16* wkT = (__bf16*)alloc((size_t)HIDDEN * EMB * 2);
  __bf16* wvT = (__bf16*)alloc((size_t)HIDDEN * EMB * 2);
  __bf16* woT = (__bf16*)alloc((size_t)EMB * HIDDEN * 2);
  __bf16* Qbf = (__bf16*)alloc((size_t)BATCH * NHEADS * SEQ * HEADDIM * 2);
  __bf16* Kbf = (__bf16*)alloc((size_t)NPREV * BATCH * NHEADS * SEQ * HEADDIM * 2);
  __bf16* Vtbf = (__bf16*)alloc((size_t)NPREV * BATCH * NHEADS * SEQ * HEADDIM * 2);
  __bf16* comb = (__bf16*)alloc((size_t)BATCH * SEQ * EMB * 2);

  // weight transpose + bf16 convert
  transpose_bf16_kernel<<<4096, 256, 0, stream>>>(wq, wqT);
  transpose_bf16_kernel<<<4096, 256, 0, stream>>>(wk, wkT);
  transpose_bf16_kernel<<<4096, 256, 0, stream>>>(wv, wvT);
  transpose_bf16_kernel<<<4096, 256, 0, stream>>>(wo, woT);

  // gating
  pool_kernel<<<(BATCH * HIDDEN) / 256, 256, 0, stream>>>(hidden, pooled);
  gate_kernel<<<1, 512, 0, stream>>>(pooled, wg, layer_w);

  // projections (WMMA GEMMs), RoPE fused for Q/K, V written transposed
  proj_gemm<0, 1, 1><<<dim3(EMB / 64, (BATCH * SEQ) / 128), 256, 0, stream>>>(
      hidden, BATCH * SEQ, wqT, bq, Qbf);
  proj_gemm<0, 1, 1><<<dim3(EMB / 64, (NPREV * BATCH * SEQ) / 128), 256, 0, stream>>>(
      prev, NPREV * BATCH * SEQ, wkT, bk, Kbf);
  proj_gemm<0, 2, 0><<<dim3(EMB / 64, (NPREV * BATCH * SEQ) / 128), 256, 0, stream>>>(
      prev, NPREV * BATCH * SEQ, wvT, bv, Vtbf);

  // fused flash attention + gating
  attention_kernel<<<dim3(SEQ / 128, NHEADS, BATCH), 256, 0, stream>>>(
      Qbf, Kbf, Vtbf, layer_w, comb);

  // output projection -> f32 d_out [B,S,HIDDEN]
  proj_gemm<1, 0, 0><<<dim3(HIDDEN / 64, (BATCH * SEQ) / 128), 256, 0, stream>>>(
      comb, BATCH * SEQ, woT, bo, (float*)d_out);
}
